// RelativePositionMultiHeadAttention_66005057405079
// MI455X (gfx1250) — compile-verified, matrix operand order checked by reading, not run on hardware
//
#include <hip/hip_runtime.h>
#include <hip/hip_bf16.h>

#define P 64
#define D 384
#define TD 1152
#define NH 12
#define THREADS 512

typedef __attribute__((ext_vector_type(16))) __bf16 bf16x16;
typedef __attribute__((ext_vector_type(8)))  float  f32x8;

__device__ __forceinline__ __bf16 f2bf(float f) {
  union { __hip_bfloat16 h; __bf16 b; } u;
  u.h = __float2bfloat16(f);
  return u.b;
}

__device__ __forceinline__ f32x8 wmma_bf(bf16x16 a, bf16x16 b, f32x8 c) {
  return __builtin_amdgcn_wmma_f32_16x16x32_bf16(false, a, false, b, (short)0, c,
                                                 false, false);
}

// Fragment load from a row-major source. Used for A tiles (lane%16 = M row) and,
// by the symmetric convention, for B tiles when the source is stored as N x K
// (lane%16 = N column). CDNA5 K striping: lanes 0-15 hold K {0..7,16..23},
// lanes 16-31 hold K {8..15,24..31}. Two aligned 16-byte chunks per lane.
__device__ __forceinline__ bf16x16 frag_ld(const __bf16* base, int stride, int row0,
                                           int col0, int lane) {
  const int lr = lane & 15, lg = lane >> 4;
  const __bf16* p = base + (row0 + lr) * stride + col0 + lg * 8;
  union { int4 q[2]; bf16x16 v; } u;
  u.q[0] = *(const int4*)(p);
  u.q[1] = *(const int4*)(p + 16);
  return u.v;
}

// ---------------------------------------------------------------------------
// Pack fp32 weights into bf16 WMMA-B fragment order.
// qkv_w: 384x1152 -> 12 kblocks x 72 ntiles; merge_w: 384x384 -> 12 x 24.
// Fragment element (lane,i): n = nt*16 + (lane&15),
//   k = kb*32 + (lane>>4)*8 + (i<8 ? i : 8+i).
// ---------------------------------------------------------------------------
__global__ __launch_bounds__(THREADS) void pack_weights(const float* __restrict__ qkv_w,
                                                        const float* __restrict__ merge_w,
                                                        __bf16* __restrict__ wq,
                                                        __bf16* __restrict__ wm) {
  const int f = blockIdx.x;            // 0..1151
  const int t = threadIdx.x;           // 0..511
  const int lane = t >> 4, i = t & 15;
  const int lr = lane & 15, lg = lane >> 4;
  const int kl = lg * 8 + i + ((i < 8) ? 0 : 8);
  if (f < 864) {
    const int kb = f / 72, nt = f % 72;
    wq[f * 512 + t] = f2bf(qkv_w[(kb * 32 + kl) * TD + nt * 16 + lr]);
  } else {
    const int f2 = f - 864;
    const int kb = f2 / 24, nt = f2 % 24;
    wm[f2 * 512 + t] = f2bf(merge_w[(kb * 32 + kl) * D + nt * 16 + lr]);
  }
}

// ---------------------------------------------------------------------------
// Fused window attention: one workgroup (16 waves) per (b,g) window.
// LDS map (245760 B total):
//   [0      , 49152 )  xb  64x384 bf16   (stage 1-2)  -- reused as outb (stage 3c-4)
//   [49152  , 147456)  qk  64x768 bf16   (q scaled | k)
//   [147456 , 196608)  vT  384x64 bf16   (v transposed: [dim][token])
//   [196608 , 229376)  sc  2 x 64x64 f32 (score tiles, head pair)
//   [229376 , 245760)  pb  2 x 64x64 bf16 (softmax probs, head pair)
// ---------------------------------------------------------------------------
__global__ __launch_bounds__(THREADS) void rpmha_main(const float* __restrict__ x,
                                                      const float* __restrict__ qkv_b,
                                                      const float* __restrict__ merge_b,
                                                      const float* __restrict__ bias_table,
                                                      const int*   __restrict__ rel_index,
                                                      const __bf16* __restrict__ wq,
                                                      const __bf16* __restrict__ wm,
                                                      float* __restrict__ out) {
  const int bg   = blockIdx.x;
  const int tid  = threadIdx.x;
  const int lane = tid & 31;
  const int wv   = tid >> 5;                 // 0..15
  const int lr   = lane & 15, lg = lane >> 4;

  __shared__ __align__(16) unsigned char smem[245760];
  __bf16* xb   = (__bf16*)smem;
  __bf16* outb = (__bf16*)smem;
  __bf16* qk   = (__bf16*)(smem + 49152);
  __bf16* vT   = (__bf16*)(smem + 147456);
  float*  sc   = (float*)(smem + 196608);
  __bf16* pb   = (__bf16*)(smem + 229376);

  // ---- Stage 1: load x tile -> LDS bf16 ----
  const float* xp = x + (size_t)bg * (P * D);
  for (int i = tid; i < P * D / 4; i += THREADS) {
    float4 v = ((const float4*)xp)[i];
    xb[4 * i + 0] = f2bf(v.x);
    xb[4 * i + 1] = f2bf(v.y);
    xb[4 * i + 2] = f2bf(v.z);
    xb[4 * i + 3] = f2bf(v.w);
  }

  // cache relative-position indices (constant across heads) in registers
  const int srow = tid >> 3, spart = tid & 7;
  int ri[8];
#pragma unroll
  for (int c = 0; c < 8; ++c)
    ri[c] = rel_index[srow * 64 + spart * 8 + c] * NH;

  __syncthreads();

  // ---- Stage 2: QKV GEMM: [64x384] @ [384x1152] + bias ----
  // q (nt<24): scaled by 1/sqrt(D), stored to qk cols 0..383
  // k (24<=nt<48): stored to qk cols 384..767
  // v (nt>=48): stored transposed into vT[dim][token]
  const float scale = 0.051031036307982884f;   // 1/sqrt(384)
  for (int t = wv; t < 4 * 72; t += 16) {
    const int mt = t / 72, nt = t % 72;
    f32x8 acc = {};
#pragma unroll
    for (int kb = 0; kb < 12; ++kb) {
      bf16x16 a = frag_ld(xb, D, mt * 16, kb * 32, lane);
      bf16x16 b = *(const bf16x16*)(wq + ((size_t)(kb * 72 + nt) * 512 + lane * 16));
      acc = wmma_bf(a, b, acc);
    }
    const int col  = nt * 16 + lr;
    const float bias = qkv_b[col];
    const int row0 = mt * 16 + lg * 8;
    if (nt < 24) {                         // q, scaled
#pragma unroll
      for (int r = 0; r < 8; ++r)
        qk[(row0 + r) * 768 + col] = f2bf((acc[r] + bias) * scale);
    } else if (nt < 48) {                  // k
#pragma unroll
      for (int r = 0; r < 8; ++r)
        qk[(row0 + r) * 768 + col] = f2bf(acc[r] + bias);
    } else {                               // v -> transposed
      const int dcol = col - 768;
#pragma unroll
      for (int r = 0; r < 8; ++r)
        vT[dcol * 64 + row0 + r] = f2bf(acc[r] + bias);
    }
  }
  __syncthreads();

  // ---- Stage 3: attention, two heads per pass ----
  for (int hp = 0; hp < NH / 2; ++hp) {
    // 3a: scores = q @ k^T : 32 tiles (2 heads x 4x4), 2 per wave
    {
#pragma unroll
      for (int half = 0; half < 2; ++half) {
        const int tt = wv + half * 16;
        const int hh = tt >> 4, mt = (tt >> 2) & 3, nt = tt & 3;
        const int h = hp * 2 + hh;
        bf16x16 a = frag_ld(qk, 768, mt * 16, h * 32, lane);        // q rows
        bf16x16 b = frag_ld(qk, 768, nt * 16, D + h * 32, lane);    // k rows as B cols
        f32x8 acc = {};
        acc = wmma_bf(a, b, acc);
        float* scp = sc + hh * 4096;
#pragma unroll
        for (int r = 0; r < 8; ++r)
          scp[(mt * 16 + lg * 8 + r) * 64 + nt * 16 + lr] = acc[r];
      }
    }
    __syncthreads();

    // 3b: softmax(scores + pos_bias) for both heads. 8 threads per row.
    {
#pragma unroll
      for (int hh = 0; hh < 2; ++hh) {
        const int h = hp * 2 + hh;
        const float* scp = sc + hh * 4096;
        __bf16* pbp = pb + hh * 4096;
        float vals[8];
        float m = -1e30f;
#pragma unroll
        for (int c = 0; c < 8; ++c) {
          float v = scp[srow * 64 + spart * 8 + c] + bias_table[ri[c] + h];
          vals[c] = v;
          m = fmaxf(m, v);
        }
        m = fmaxf(m, __shfl_xor(m, 1, 32));
        m = fmaxf(m, __shfl_xor(m, 2, 32));
        m = fmaxf(m, __shfl_xor(m, 4, 32));
        float s = 0.f;
#pragma unroll
        for (int c = 0; c < 8; ++c) { vals[c] = __expf(vals[c] - m); s += vals[c]; }
        s += __shfl_xor(s, 1, 32);
        s += __shfl_xor(s, 2, 32);
        s += __shfl_xor(s, 4, 32);
        const float inv = 1.0f / s;
#pragma unroll
        for (int c = 0; c < 8; ++c)
          pbp[srow * 64 + spart * 8 + c] = f2bf(vals[c] * inv);
      }
    }
    __syncthreads();

    // 3c: out_h = attn @ v : 16 tiles (2 heads x 4x2), 1 per wave, K=64 -> 2 WMMAs
    {
      const int hh = wv >> 3, mt = (wv >> 1) & 3, dt = wv & 1;
      const int h = hp * 2 + hh;
      const int dbase = h * 32 + dt * 16;
      const __bf16* pbp = pb + hh * 4096;
      f32x8 acc = {};
#pragma unroll
      for (int kb = 0; kb < 2; ++kb) {
        bf16x16 a = frag_ld(pbp, 64, mt * 16, kb * 32, lane);
        bf16x16 b = frag_ld(vT, 64, dbase, kb * 32, lane);   // vT[dim][token]
        acc = wmma_bf(a, b, acc);
      }
#pragma unroll
      for (int r = 0; r < 8; ++r)
        outb[(mt * 16 + lg * 8 + r) * D + dbase + lr] = f2bf(acc[r]);
    }
    __syncthreads();
  }

  // ---- Stage 4: merge GEMM: [64x384] @ [384x384] + bias -> global fp32 ----
  float* op = out + (size_t)bg * (P * D);
  for (int t = wv; t < 4 * 24; t += 16) {
    const int mt = t / 24, nt = t % 24;
    f32x8 acc = {};
#pragma unroll
    for (int kb = 0; kb < 12; ++kb) {
      bf16x16 a = frag_ld(outb, D, mt * 16, kb * 32, lane);
      bf16x16 b = *(const bf16x16*)(wm + ((size_t)(kb * 24 + nt) * 512 + lane * 16));
      acc = wmma_bf(a, b, acc);
    }
    const int col = nt * 16 + lr;
    const float mb = merge_b[col];
#pragma unroll
    for (int r = 0; r < 8; ++r)
      op[(mt * 16 + lg * 8 + r) * D + col] = acc[r] + mb;
  }
}

extern "C" void kernel_launch(void* const* d_in, const int* in_sizes, int n_in,
                              void* d_out, int out_size, void* d_ws, size_t ws_size,
                              hipStream_t stream) {
  const float* x          = (const float*)d_in[0];
  const float* qkv_w      = (const float*)d_in[1];
  const float* qkv_b      = (const float*)d_in[2];
  const float* merge_w    = (const float*)d_in[3];
  const float* merge_b    = (const float*)d_in[4];
  const float* bias_table = (const float*)d_in[5];
  const int*   rel_index  = (const int*)d_in[6];
  float* out = (float*)d_out;

  __bf16* wq = (__bf16*)d_ws;                  // 864 frags * 512 bf16
  __bf16* wm = wq + (size_t)864 * 512;         // 288 frags * 512 bf16

  pack_weights<<<1152, THREADS, 0, stream>>>(qkv_w, merge_w, wq, wm);
  rpmha_main<<<2048, THREADS, 0, stream>>>(x, qkv_b, merge_b, bias_table, rel_index,
                                           wq, wm, out);
}
